// Block_15556371546952
// MI455X (gfx1250) — matrix operand order repaired and checked by reading, not compiled, not run on hardware
//
#include <hip/hip_runtime.h>
#include <cstdint>
#include <cstddef>

// ---------------------------------------------------------------------------
// Types for CDNA5 WMMA (gfx1250, wave32)
// ---------------------------------------------------------------------------
typedef __bf16 bf16;
typedef __attribute__((ext_vector_type(16))) __bf16 v16bf;
typedef __attribute__((ext_vector_type(8)))  float  v8f;
typedef __attribute__((ext_vector_type(4)))  unsigned int cdna5_u32x4;
typedef __attribute__((ext_vector_type(8)))  int          cdna5_i32x8;
typedef __attribute__((ext_vector_type(4)))  int          cdna5_i32x4;

#ifndef __has_builtin
#define __has_builtin(x) 0
#endif

// Tensor Data Mover availability (device pass only; host pass uses fallback)
#if defined(__HIP_DEVICE_COMPILE__) && \
    __has_builtin(__builtin_amdgcn_tensor_load_to_lds) && \
    __has_builtin(__builtin_amdgcn_s_wait_tensorcnt)
#define CDNA5_HAS_TDM 1
#else
#define CDNA5_HAS_TDM 0
#endif

// therock-10.0 headers ship only with the 6-arg-builtin toolchain (clang-23)
#if __has_include(<hip/amd_detail/amd_gfx1250_TDM.h>)
#define CDNA5_TDM_6ARG 1
#else
#define CDNA5_TDM_6ARG 0
#endif

#define TSEQ 2048
#define CEMB 4096
#define QKVD 6144
#define FFND 11008

__device__ __forceinline__ bf16 f2bf(float f) {
  unsigned u = __builtin_bit_cast(unsigned, f);
  unsigned r = (u + 0x7fffu + ((u >> 16) & 1u)) >> 16;   // round-to-nearest-even
  unsigned short s = (unsigned short)r;
  return __builtin_bit_cast(bf16, s);
}

union Frag32 { v16bf v; uint4 u[2]; };

// 16 contiguous bf16 (B-matrix fragment chunk: lane%16 = N column, lane/16 = K half)
__device__ __forceinline__ v16bf ld_contig16(const bf16* p) {
  Frag32 f;
  f.u[0] = *(const uint4*)(p);
  f.u[1] = *(const uint4*)(p + 8);
  return f.v;
}
// A-matrix fragment: two 8-half chunks at +0 and +8 halfs (caller adds (lane>>4)*16)
__device__ __forceinline__ v16bf ld_a_frag(const bf16* p) {
  Frag32 f;
  f.u[0] = *(const uint4*)(p);
  f.u[1] = *(const uint4*)(p + 8);
  return f.v;
}

__device__ __forceinline__ v8f wmma_bf16(v16bf a, v16bf b, v8f c) {
  // v_wmma_f32_16x16x32_bf16  (8-arg form: neg_a, A, neg_b, B, c_mod, C, reuse_a, reuse_b)
  return __builtin_amdgcn_wmma_f32_16x16x32_bf16(false, a, false, b, (short)0, c,
                                                 false, false);
}

#if CDNA5_HAS_TDM
// Low 32 bits of an LDS flat address are the LDS byte offset (aperture in MSBs).
__device__ __forceinline__ unsigned lds_byte_addr(const void* p) {
  return (unsigned)(size_t)p;
}

// Issue one TDM 2D tile load: 128 rows x 32 bf16 from a row-major [*,K] matrix
// (row stride = K elements) into LDS, packed row-major (row stride 32 halfs).
// D# per CDNA5 ISA ch.8: group0 = {count/lds_addr/global_addr/type},
// group1 = {data_size, tensor dims, tile dims, dim0 stride}. 2D => groups 2/3 zero.
__device__ __forceinline__ void tdm_load_tile_128x32(unsigned lds_addr,
                                                     const bf16* gsrc, int K) {
  const unsigned long long ga = (unsigned long long)(size_t)gsrc;
  cdna5_u32x4 g0;
  g0[0] = 1u;                                                // count=1 (valid user D#)
  g0[1] = lds_addr;                                          // lds_addr (bytes)
  g0[2] = (unsigned)ga;                                      // global_addr[31:0]
  g0[3] = (unsigned)((ga >> 32) & 0x01ffffffu) | (2u << 30); // global_addr[56:32] | type=2
  cdna5_i32x8 g1;
  g1[0] = (int)(1u << 16);                                   // data_size=1 (2 bytes)
  g1[1] = (int)(((unsigned)K & 0xffffu) << 16);              // tensor_dim0[15:0]
  g1[2] = (int)((((unsigned)K >> 16) & 0xffffu) | (128u << 16)); // dim0[31:16] | dim1[15:0]
  g1[3] = (int)(32u << 16);                                  // dim1[31:16]=0 | tile_dim0=32
  g1[4] = (int)128u;                                         // tile_dim1=128, tile_dim2=0
  g1[5] = (int)(unsigned)K;                                  // tensor_dim0_stride[31:0]
  g1[6] = 0;                                                 // stride[47:32] | dim1_stride lo
  g1[7] = 0;                                                 // dim1_stride hi
  cdna5_i32x4 z4 = {0, 0, 0, 0};
#if CDNA5_TDM_6ARG
  cdna5_i32x8 z8 = {0, 0, 0, 0, 0, 0, 0, 0};
  __builtin_amdgcn_tensor_load_to_lds(g0, g1, z4, z4, z8, 0);
#else
  __builtin_amdgcn_tensor_load_to_lds(g0, g1, z4, z4, 0);
#endif
}
#endif  // CDNA5_HAS_TDM

// ---------------------------------------------------------------------------
// fp32 -> bf16 conversion (weights / activations)
// ---------------------------------------------------------------------------
__global__ void cvt_f32_bf16(const float* __restrict__ in, bf16* __restrict__ out,
                             size_t n) {
  size_t i = (size_t)blockIdx.x * blockDim.x + threadIdx.x;
  size_t stride = (size_t)gridDim.x * blockDim.x;
  for (; i < n; i += stride) out[i] = f2bf(in[i]);
}

// ---------------------------------------------------------------------------
// RMSNorm: per-row, output bf16
// ---------------------------------------------------------------------------
__global__ __launch_bounds__(256) void rmsnorm_bf16(const float* __restrict__ x,
                                                    const float* __restrict__ w,
                                                    bf16* __restrict__ out, int C) {
  const int t = blockIdx.x;
  const float* xr = x + (size_t)t * C;
  float s = 0.f;
  for (int i = threadIdx.x; i < C; i += 256) { float v = xr[i]; s += v * v; }
  __shared__ float red[256];
  red[threadIdx.x] = s;
  __syncthreads();
  for (int o = 128; o > 0; o >>= 1) {
    if (threadIdx.x < (unsigned)o) red[threadIdx.x] += red[threadIdx.x + o];
    __syncthreads();
  }
  const float rs = rsqrtf(red[0] / (float)C + 1e-5f);
  bf16* orow = out + (size_t)t * C;
  for (int i = threadIdx.x; i < C; i += 256) orow[i] = f2bf(xr[i] * rs * w[i]);
}

// ---------------------------------------------------------------------------
// Generic bf16 WMMA GEMM:  C[M,N] (f32) = A[M,K] (bf16) * W[N,K]^T (bf16)
// Block tile 128x128, K-step 32, LDS double-buffered. 8 waves: 4(M) x 2(N),
// each wave computes 32x64 = 2x4 tiles of v_wmma_f32_16x16x32_bf16.
// Tile staging: Tensor Data Mover (wave 0 issues, TENSORcnt-tracked) when the
// builtin exists; otherwise manual global_load_b128 + ds_store_b128.
// M % 128 == 0, N % 128 == 0, K % 32 == 0 (true for all uses here).
// ---------------------------------------------------------------------------
__global__ __launch_bounds__(256) void gemm_bf16(const bf16* __restrict__ A,
                                                 const bf16* __restrict__ W,
                                                 float* __restrict__ C,
                                                 int M, int N, int K) {
  __shared__ bf16 sA[2][128 * 32];
  __shared__ bf16 sB[2][128 * 32];
  const int tid = threadIdx.x, lane = tid & 31, wave = tid >> 5;
  const int wm = wave & 3, wn = wave >> 2;
  const int m0 = blockIdx.y * 128, n0 = blockIdx.x * 128;
  const bf16* Ab = A + (size_t)m0 * K;
  const bf16* Wb = W + (size_t)n0 * K;

  v8f acc[2][4] = {};
  const int kt_n = K / 32;

#if CDNA5_HAS_TDM
  // ---- TDM staging: wave 0 drives the Tensor Data Mover ----
  const bool issuer = (wave == 0);
  if (issuer) {
    tdm_load_tile_128x32(lds_byte_addr(&sA[0][0]), Ab, K);
    tdm_load_tile_128x32(lds_byte_addr(&sB[0][0]), Wb, K);
  }
  for (int kt = 0; kt < kt_n; ++kt) {
    const int buf = kt & 1;
    if (issuer) {
      if (kt + 1 < kt_n) {
        const int k1 = (kt + 1) * 32;
        tdm_load_tile_128x32(lds_byte_addr(&sA[buf ^ 1][0]), Ab + k1, K);
        tdm_load_tile_128x32(lds_byte_addr(&sB[buf ^ 1][0]), Wb + k1, K);
        __builtin_amdgcn_s_wait_tensorcnt(2);   // current buf's 2 loads done
      } else {
        __builtin_amdgcn_s_wait_tensorcnt(0);
      }
    }
    __syncthreads();   // publish TDM-written tile to all 8 waves
    const int row = lane & 15, hi = lane >> 4;
    v16bf af[2], bfg[4];
#pragma unroll
    for (int im = 0; im < 2; ++im)
      af[im] = ld_a_frag(&sA[buf][(wm * 32 + im * 16 + row) * 32 + hi * 16]);
#pragma unroll
    for (int in = 0; in < 4; ++in)
      bfg[in] = ld_contig16(&sB[buf][(wn * 64 + in * 16 + row) * 32 + hi * 16]);
#pragma unroll
    for (int im = 0; im < 2; ++im)
#pragma unroll
      for (int in = 0; in < 4; ++in)
        acc[im][in] = wmma_bf16(af[im], bfg[in], acc[im][in]);
    __syncthreads();   // all waves done with buf before TDM overwrites it
  }
#else
  // ---- Fallback staging: global_load_b128 + ds_store_b128 ----
  const int srow = tid >> 1, scol = (tid & 1) * 16;   // 256 thr x 16 halfs = 128x32
  {
    const uint4* sa = (const uint4*)(Ab + (size_t)srow * K + scol);
    uint4* da = (uint4*)(&sA[0][srow * 32 + scol]);
    da[0] = sa[0]; da[1] = sa[1];
    const uint4* sb = (const uint4*)(Wb + (size_t)srow * K + scol);
    uint4* db = (uint4*)(&sB[0][srow * 32 + scol]);
    db[0] = sb[0]; db[1] = sb[1];
  }
  __syncthreads();
  for (int kt = 0; kt < kt_n; ++kt) {
    const int buf = kt & 1;
    if (kt + 1 < kt_n) {
      const int k1 = (kt + 1) * 32;
      __builtin_prefetch(Ab + (size_t)srow * K + k1 + 32 + scol, 0, 1);
      __builtin_prefetch(Wb + (size_t)srow * K + k1 + 32 + scol, 0, 1);
      const uint4* sa = (const uint4*)(Ab + (size_t)srow * K + k1 + scol);
      uint4* da = (uint4*)(&sA[buf ^ 1][srow * 32 + scol]);
      da[0] = sa[0]; da[1] = sa[1];
      const uint4* sb = (const uint4*)(Wb + (size_t)srow * K + k1 + scol);
      uint4* db = (uint4*)(&sB[buf ^ 1][srow * 32 + scol]);
      db[0] = sb[0]; db[1] = sb[1];
    }
    const int row = lane & 15, hi = lane >> 4;
    v16bf af[2], bfg[4];
#pragma unroll
    for (int im = 0; im < 2; ++im)
      af[im] = ld_a_frag(&sA[buf][(wm * 32 + im * 16 + row) * 32 + hi * 16]);
#pragma unroll
    for (int in = 0; in < 4; ++in)
      bfg[in] = ld_contig16(&sB[buf][(wn * 64 + in * 16 + row) * 32 + hi * 16]);
#pragma unroll
    for (int im = 0; im < 2; ++im)
#pragma unroll
      for (int in = 0; in < 4; ++in)
        acc[im][in] = wmma_bf16(af[im], bfg[in], acc[im][in]);
    __syncthreads();
  }
#endif

  const int rbase = (lane >> 4) * 8, col = lane & 15;
#pragma unroll
  for (int im = 0; im < 2; ++im)
#pragma unroll
    for (int in = 0; in < 4; ++in) {
      const int mm = m0 + wm * 32 + im * 16 + rbase;
      const int nn = n0 + wn * 64 + in * 16 + col;
#pragma unroll
      for (int v = 0; v < 8; ++v)
        C[(size_t)(mm + v) * N + nn] = acc[im][in][v];
    }
}

// ---------------------------------------------------------------------------
// RoPE + QKV split: qkv f32 [T, 6144] -> q_bf [32,T,128], k_bf [8,T,128],
// v_bf [8,T,128].  blockIdx.y: 0..31 = q head, 32..39 = k group, 40..47 = v.
// ---------------------------------------------------------------------------
__global__ __launch_bounds__(128) void rope_split(const float* __restrict__ qkv,
                                                  const float* __restrict__ cosb,
                                                  const float* __restrict__ sinb,
                                                  bf16* __restrict__ qb,
                                                  bf16* __restrict__ kb,
                                                  bf16* __restrict__ vb, int T) {
  const int t = blockIdx.x, u = blockIdx.y, d = threadIdx.x;
  const float* rowp = qkv + (size_t)t * QKVD;
  const float c = cosb[(size_t)t * 128 + d];
  const float s = sinb[(size_t)t * 128 + d];
  if (u < 32) {
    const int h = u, g = h >> 2, j = h & 3;
    const float* p = rowp + (g * 6 + j) * 128;
    const float xv = p[d];
    const float rot = (d < 64) ? -p[d + 64] : p[d - 64];
    qb[((size_t)h * T + t) * 128 + d] = f2bf(xv * c + rot * s);
  } else if (u < 40) {
    const int g = u - 32;
    const float* p = rowp + (g * 6 + 4) * 128;
    const float xv = p[d];
    const float rot = (d < 64) ? -p[d + 64] : p[d - 64];
    kb[((size_t)g * T + t) * 128 + d] = f2bf(xv * c + rot * s);
  } else {
    const int g = u - 40;
    vb[((size_t)g * T + t) * 128 + d] = f2bf(rowp[(g * 6 + 5) * 128 + d]);
  }
}

// relu(n1@scale_w^T + b) -> per-head mean over 128 channels -> scale_t [32, T]
__global__ void scale_reduce(const float* __restrict__ sc,
                             const float* __restrict__ bias,
                             float* __restrict__ st, int T) {
  const int idx = blockIdx.x * blockDim.x + threadIdx.x;
  if (idx >= T * 32) return;
  const int t = idx >> 5, h = idx & 31;
  const float* p = sc + (size_t)t * CEMB + h * 128;
  const float* bp = bias + h * 128;
  float s = 0.f;
  for (int d = 0; d < 128; ++d) { float v = p[d] + bp[d]; s += fmaxf(v, 0.f); }
  st[(size_t)h * T + t] = s * (1.0f / 128.0f);
}

// ---------------------------------------------------------------------------
// Flash attention (non-causal), WMMA for QK^T and PV.
// Block = (head, 128 q-rows), 8 waves x 16 q-rows. KV tiles of 32 in LDS.
// Online softmax in f32 accumulators (lane = col, VGPR = row layout), row
// reductions via wave32 shfl_xor inside 16-lane halves.
// ---------------------------------------------------------------------------
__global__ __launch_bounds__(256) void attn_fa(const bf16* __restrict__ qb,
                                               const bf16* __restrict__ kb,
                                               const bf16* __restrict__ vb,
                                               const float* __restrict__ scl,
                                               bf16* __restrict__ yb, int T) {
  __shared__ bf16 sK[32 * 128];    // [kk][d]
  __shared__ bf16 sVt[128 * 32];   // [d][kk] (transposed)
  __shared__ bf16 sP[8][16 * 32];  // per-wave P re-swizzle scratch
  const int h = blockIdx.y, g = h >> 2;
  const int q0 = blockIdx.x * 128;
  const int tid = threadIdx.x, lane = tid & 31, wave = tid >> 5;
  const int qr = q0 + wave * 16;
  const int row = lane & 15, hi = lane >> 4, rbase = hi * 8;

  // Persistent Q fragments: 4 k-steps over head dim 128
  v16bf qf[4];
  {
    const bf16* qp = qb + ((size_t)h * T + qr + row) * 128;
#pragma unroll
    for (int s = 0; s < 4; ++s) qf[s] = ld_a_frag(qp + s * 32 + hi * 16);
  }
  float rs[8];
#pragma unroll
  for (int v = 0; v < 8; ++v)
    rs[v] = scl[(size_t)h * T + qr + rbase + v] * 0.0883883476483184f; // 1/sqrt(128)

  float mi[8], li[8];
  v8f yacc[8] = {};
#pragma unroll
  for (int v = 0; v < 8; ++v) { mi[v] = -1e30f; li[v] = 0.f; }

  const int srowK = tid >> 3, scolK = (tid & 7) * 16;  // 256 thr stage 32x128
  for (int kt = 0; kt < T / 32; ++kt) {
    const int kk0 = kt * 32;
    {
      const uint4* src = (const uint4*)(kb + ((size_t)g * T + kk0 + srowK) * 128 + scolK);
      uint4* dst = (uint4*)(&sK[srowK * 128 + scolK]);
      dst[0] = src[0]; dst[1] = src[1];
      const bf16* vsrc = vb + ((size_t)g * T + kk0 + srowK) * 128 + scolK;
#pragma unroll
      for (int j = 0; j < 16; ++j) sVt[(scolK + j) * 32 + srowK] = vsrc[j];
    }
    __syncthreads();

    // S tile: 16(q) x 32(kk), K-dim = 128 head dims in 4 steps
    v8f sacc[2] = {};
#pragma unroll
    for (int s = 0; s < 4; ++s)
#pragma unroll
      for (int nt = 0; nt < 2; ++nt) {
        v16bf kf = ld_contig16(&sK[(nt * 16 + row) * 128 + s * 32 + hi * 16]);
        sacc[nt] = wmma_bf16(qf[s], kf, sacc[nt]);
      }

    float sm[8];
#pragma unroll
    for (int v = 0; v < 8; ++v) {
      const float a = sacc[0][v] * rs[v];
      const float b = sacc[1][v] * rs[v];
      sacc[0][v] = a; sacc[1][v] = b;
      sm[v] = fmaxf(a, b);
    }
#pragma unroll
    for (int off = 1; off <= 8; off <<= 1)
#pragma unroll
      for (int v = 0; v < 8; ++v) sm[v] = fmaxf(sm[v], __shfl_xor(sm[v], off));

    float ps[8];
#pragma unroll
    for (int v = 0; v < 8; ++v) {
      const float mn = fmaxf(mi[v], sm[v]);
      const float alpha = __expf(mi[v] - mn);
      mi[v] = mn;
      const float p0 = __expf(sacc[0][v] - mn);
      const float p1 = __expf(sacc[1][v] - mn);
      sP[wave][(rbase + v) * 32 + row] = f2bf(p0);
      sP[wave][(rbase + v) * 32 + 16 + row] = f2bf(p1);
      ps[v] = p0 + p1;
      li[v] *= alpha;
#pragma unroll
      for (int dt = 0; dt < 8; ++dt) yacc[dt][v] *= alpha;
    }
#pragma unroll
    for (int off = 1; off <= 8; off <<= 1)
#pragma unroll
      for (int v = 0; v < 8; ++v) ps[v] += __shfl_xor(ps[v], off);
#pragma unroll
    for (int v = 0; v < 8; ++v) li[v] += ps[v];

    // PV: A = P (16x32), B = V (32x16 per d-tile), same-wave LDS RAW is in-order
    v16bf pf = ld_a_frag(&sP[wave][row * 32 + hi * 16]);
#pragma unroll
    for (int dt = 0; dt < 8; ++dt) {
      v16bf vf = ld_contig16(&sVt[(dt * 16 + row) * 32 + hi * 16]);
      yacc[dt] = wmma_bf16(pf, vf, yacc[dt]);
    }
    __syncthreads();
  }

#pragma unroll
  for (int v = 0; v < 8; ++v) {
    const float inv = 1.0f / li[v];
    const size_t orow = (size_t)(qr + rbase + v) * CEMB + (size_t)h * 128;
#pragma unroll
    for (int dt = 0; dt < 8; ++dt)
      yb[orow + dt * 16 + row] = f2bf(yacc[dt][v] * inv);
  }
}

// ---------------------------------------------------------------------------
// Elementwise epilogues
// ---------------------------------------------------------------------------
__global__ void add_f32(const float* __restrict__ a, const float* __restrict__ b,
                        float* __restrict__ o, size_t n) {
  size_t i = (size_t)blockIdx.x * blockDim.x + threadIdx.x;
  size_t stride = (size_t)gridDim.x * blockDim.x;
  for (; i < n; i += stride) o[i] = a[i] + b[i];
}

__global__ void silu_mul_bf16(const float* __restrict__ g, const float* __restrict__ u,
                              bf16* __restrict__ o, size_t n) {
  size_t i = (size_t)blockIdx.x * blockDim.x + threadIdx.x;
  size_t stride = (size_t)gridDim.x * blockDim.x;
  for (; i < n; i += stride) {
    const float gv = g[i];
    const float s = gv / (1.f + __expf(-gv));
    o[i] = f2bf(s * u[i]);
  }
}

// ---------------------------------------------------------------------------
// Launcher
// ---------------------------------------------------------------------------
extern "C" void kernel_launch(void* const* d_in, const int* in_sizes, int n_in,
                              void* d_out, int out_size, void* d_ws, size_t ws_size,
                              hipStream_t stream) {
  (void)in_sizes; (void)n_in; (void)out_size; (void)ws_size;
  const float* x       = (const float*)d_in[0];
  const float* cosb    = (const float*)d_in[1];
  const float* sinb    = (const float*)d_in[2];
  const float* norm1_w = (const float*)d_in[3];
  const float* norm2_w = (const float*)d_in[4];
  const float* attn_w  = (const float*)d_in[5];
  const float* proj_w  = (const float*)d_in[6];
  const float* scale_w = (const float*)d_in[7];
  const float* scale_b = (const float*)d_in[8];
  const float* gate_w  = (const float*)d_in[9];
  const float* up_w    = (const float*)d_in[10];
  const float* down_w  = (const float*)d_in[11];
  float* out = (float*)d_out;

  const size_t T = TSEQ, C = CEMB, Q = QKVD, F = FFND;
  char* ws = (char*)d_ws;
  size_t off = 0;
  auto alloc = [&](size_t bytes) -> void* {
    void* p = ws + off;
    off = (off + bytes + 255) & ~(size_t)255;
    return p;
  };
  bf16* w_attn  = (bf16*)alloc(Q * C * 2);
  bf16* w_scale = (bf16*)alloc(C * C * 2);
  bf16* w_proj  = (bf16*)alloc(C * C * 2);
  bf16* w_gate  = (bf16*)alloc(F * C * 2);
  bf16* w_up    = (bf16*)alloc(F * C * 2);
  bf16* w_down  = (bf16*)alloc(C * F * 2);
  bf16* n1b     = (bf16*)alloc(T * C * 2);
  bf16* n2b     = (bf16*)alloc(T * C * 2);
  bf16* qbuf    = (bf16*)alloc(32 * T * 128 * 2);
  bf16* kbuf    = (bf16*)alloc(8 * T * 128 * 2);
  bf16* vbuf    = (bf16*)alloc(8 * T * 128 * 2);
  bf16* ybuf    = (bf16*)alloc(T * C * 2);
  bf16* mbuf    = (bf16*)alloc(T * F * 2);
  float* qkvf   = (float*)alloc(T * Q * 4);
  float* scf    = (float*)alloc(T * C * 4);
  float* sclt   = (float*)alloc(32 * T * 4);
  float* hbuf   = (float*)alloc(T * C * 4);
  float* x1     = (float*)alloc(T * C * 4);
  float* gatef  = (float*)alloc(T * F * 4);
  float* upf    = (float*)alloc(T * F * 4);
  float* dbuf   = (float*)alloc(T * C * 4);

  // 1) weights -> bf16
  cvt_f32_bf16<<<2048, 256, 0, stream>>>(attn_w,  w_attn,  Q * C);
  cvt_f32_bf16<<<2048, 256, 0, stream>>>(scale_w, w_scale, C * C);
  cvt_f32_bf16<<<2048, 256, 0, stream>>>(proj_w,  w_proj,  C * C);
  cvt_f32_bf16<<<2048, 256, 0, stream>>>(gate_w,  w_gate,  F * C);
  cvt_f32_bf16<<<2048, 256, 0, stream>>>(up_w,    w_up,    F * C);
  cvt_f32_bf16<<<2048, 256, 0, stream>>>(down_w,  w_down,  C * F);

  // 2) n1 = rmsnorm(x)
  rmsnorm_bf16<<<(int)T, 256, 0, stream>>>(x, norm1_w, n1b, (int)C);

  // 3) qkv = n1 @ attn_w^T      [2048 x 6144]
  gemm_bf16<<<dim3(Q / 128, T / 128), 256, 0, stream>>>(n1b, w_attn, qkvf,
                                                        (int)T, (int)Q, (int)C);
  // 4) scaling = n1 @ scale_w^T [2048 x 4096], then relu+bias+head-mean
  gemm_bf16<<<dim3(C / 128, T / 128), 256, 0, stream>>>(n1b, w_scale, scf,
                                                        (int)T, (int)C, (int)C);
  scale_reduce<<<(int)(T * 32 / 256), 256, 0, stream>>>(scf, scale_b, sclt, (int)T);

  // 5) RoPE + split into q/k/v bf16
  rope_split<<<dim3(T, 48), 128, 0, stream>>>(qkvf, cosb, sinb, qbuf, kbuf, vbuf,
                                              (int)T);

  // 6) flash attention
  attn_fa<<<dim3(T / 128, 32), 256, 0, stream>>>(qbuf, kbuf, vbuf, sclt, ybuf,
                                                 (int)T);

  // 7) h = y @ proj_w^T ; x1 = x + h
  gemm_bf16<<<dim3(C / 128, T / 128), 256, 0, stream>>>(ybuf, w_proj, hbuf,
                                                        (int)T, (int)C, (int)C);
  add_f32<<<2048, 256, 0, stream>>>(x, hbuf, x1, T * C);

  // 8) n2 = rmsnorm(x1)
  rmsnorm_bf16<<<(int)T, 256, 0, stream>>>(x1, norm2_w, n2b, (int)C);

  // 9) MLP: gate/up GEMMs, silu*up, down GEMM
  gemm_bf16<<<dim3(F / 128, T / 128), 256, 0, stream>>>(n2b, w_gate, gatef,
                                                        (int)T, (int)F, (int)C);
  gemm_bf16<<<dim3(F / 128, T / 128), 256, 0, stream>>>(n2b, w_up, upf,
                                                        (int)T, (int)F, (int)C);
  silu_mul_bf16<<<2048, 256, 0, stream>>>(gatef, upf, mbuf, T * F);
  gemm_bf16<<<dim3(C / 128, T / 128), 256, 0, stream>>>(mbuf, w_down, dbuf,
                                                        (int)T, (int)C, (int)F);

  // 10) out = x1 + down
  add_f32<<<2048, 256, 0, stream>>>(x1, dbuf, out, T * C);
}
